// CustomAttention_27659589386683
// MI455X (gfx1250) — matrix operand order repaired and checked by reading, not compiled, not run on hardware
//
#include <hip/hip_runtime.h>
#include <stdint.h>

#define NUM_HEADS 12
#define HEAD_DIM  64
#define LN_EPS    1e-5f

typedef __attribute__((ext_vector_type(16))) __bf16 v16bf;
typedef __attribute__((ext_vector_type(8)))  float  v8f;
typedef unsigned short u16;   // bf16 storage in memory / LDS

// ---------------------------------------------------------------- helpers

__device__ __forceinline__ u16 f2bf(float f) {
    union { float f; unsigned int u; } c; c.f = f;
    unsigned int u = c.u;
    unsigned int r = u + 0x7FFFu + ((u >> 16) & 1u);   // round-nearest-even
    return (u16)(r >> 16);
}

__device__ __forceinline__ v16bf frag_from_2x16B(const u16* p0, const u16* p1) {
    union { uint4 q[2]; v16bf v; } t;
    t.q[0] = *(const uint4*)p0;
    t.q[1] = *(const uint4*)p1;
    return t.v;
}
__device__ __forceinline__ v16bf frag_from_32B(const u16* p) {
    return frag_from_2x16B(p, p + 8);
}

// ------------------------------------------------- async global -> LDS copy
// CDNA5 GLOBAL_LOAD_ASYNC_TO_LDS_B128 (ASYNCcnt-tracked); falls back to a
// synchronous VGPR round-trip if the builtin is not available.
// Builtin signature (from hipcc diagnostic): (v4i AS1*, v4i AS3*, Ii, Ii).

#if defined(__gfx1250__) && __has_builtin(__builtin_amdgcn_global_load_async_to_lds_b128) && __has_builtin(__builtin_amdgcn_s_wait_asynccnt)
#define USE_ASYNC_LDS 1
#else
#define USE_ASYNC_LDS 0
#endif

#if USE_ASYNC_LDS
typedef int v4i_b128 __attribute__((vector_size(16)));
typedef __attribute__((address_space(1))) v4i_b128 as1_v4i;
typedef __attribute__((address_space(3))) v4i_b128 as3_v4i;
#endif

__device__ __forceinline__ void g2lds_16B(const u16* gsrc, u16* ldst) {
#if USE_ASYNC_LDS
    __builtin_amdgcn_global_load_async_to_lds_b128(
        (as1_v4i*)(unsigned long long)(const void*)gsrc,
        (as3_v4i*)(unsigned long long)(const void*)ldst, 0, 0);
#else
    *(uint4*)ldst = *(const uint4*)gsrc;
#endif
}
__device__ __forceinline__ void wait_g2lds() {
#if USE_ASYNC_LDS
    __builtin_amdgcn_s_wait_asynccnt(0);
#endif
}

// ---------------------------------------------------------------- fp32 -> bf16

__global__ void cvt_f32_bf16(const float* __restrict__ src, u16* __restrict__ dst, int n) {
    int i = blockIdx.x * blockDim.x + threadIdx.x;
    int stride = gridDim.x * blockDim.x;
    for (; i < n; i += stride) dst[i] = f2bf(src[i]);
}

// ---------------------------------------------------------------- WMMA GEMM
// Out[m, o] = sum_k A[m,k] * W[o,k] (+ bias[o]).  A:[M,K] bf16, W:[Nout,K] bf16.
// Block: 256 threads = 8 waves (2 x 4).  Block tile 128(M) x 256(N), K-step 32.
// Wave tile 64x64 = 4x4 fragments -> 16 WMMAs per 8 fragment loads.
// LDS double-buffered, next K-slab async-copied during compute.

__global__ __launch_bounds__(256)
void gemm_bf16_wmma(const u16* __restrict__ A, const u16* __restrict__ W,
                    const float* __restrict__ bias, float* __restrict__ Out,
                    int M, int K, int Nout)
{
    __shared__ __align__(16) u16 Ashm[2][8][32][16];   // [buf][mblk][lane][e]
    __shared__ __align__(16) u16 Bshm[2][16][32][16];  // [buf][nblk][lane][e]

    const int tid   = threadIdx.x;
    const int lane  = tid & 31;
    const int wid   = tid >> 5;       // 0..7
    const int waveM = wid >> 2;       // 0..1  (64 rows each)
    const int waveN = wid & 3;        // 0..3  (64 cols each)
    const int ln16  = lane & 15;
    const int hi    = (lane >> 4) & 1;

    const int rowBase = blockIdx.y * 128;
    const int colBase = blockIdx.x * 256;

    const v8f vzero = {0.f,0.f,0.f,0.f,0.f,0.f,0.f,0.f};
    v8f acc[4][4];
    #pragma unroll
    for (int i = 0; i < 4; ++i)
        #pragma unroll
        for (int j = 0; j < 4; ++j) acc[i][j] = vzero;

    // Loader slots: this thread fills A-blk wid and B-blks {2*wid, 2*wid+1}.
    const int aRow  = rowBase + wid * 16 + ln16;
    const int bRow0 = colBase + (wid * 2 + 0) * 16 + ln16;
    const int bRow1 = colBase + (wid * 2 + 1) * 16 + ln16;

    auto issue = [&](int k0, int buf) {
        // A fragment-ready: k = k0 + hi*16 + e (16 contiguous halves)
        const u16* ga = A + (size_t)aRow * K + k0 + hi * 16;
        g2lds_16B(ga,      &Ashm[buf][wid][lane][0]);
        g2lds_16B(ga + 8,  &Ashm[buf][wid][lane][8]);
        // B fragment-ready: k = k0 + hi*16 + e (16 contiguous halves)
        const u16* gb0 = W + (size_t)bRow0 * K + k0 + hi * 16;
        g2lds_16B(gb0,     &Bshm[buf][wid * 2 + 0][lane][0]);
        g2lds_16B(gb0 + 8, &Bshm[buf][wid * 2 + 0][lane][8]);
        const u16* gb1 = W + (size_t)bRow1 * K + k0 + hi * 16;
        g2lds_16B(gb1,     &Bshm[buf][wid * 2 + 1][lane][0]);
        g2lds_16B(gb1 + 8, &Bshm[buf][wid * 2 + 1][lane][8]);
    };

    issue(0, 0);
    int buf = 0;
    for (int k0 = 0; k0 < K; k0 += 32) {
        wait_g2lds();          // own wave's async copies for `buf` done
        __syncthreads();       // everyone's copies done; prev compute retired
        if (k0 + 32 < K) issue(k0 + 32, buf ^ 1);   // overlap with WMMAs below

        v16bf a[4], bq[4];
        #pragma unroll
        for (int i = 0; i < 4; ++i) a[i]  = frag_from_32B(&Ashm[buf][waveM * 4 + i][lane][0]);
        #pragma unroll
        for (int j = 0; j < 4; ++j) bq[j] = frag_from_32B(&Bshm[buf][waveN * 4 + j][lane][0]);
        #pragma unroll
        for (int i = 0; i < 4; ++i)
            #pragma unroll
            for (int j = 0; j < 4; ++j)
                acc[i][j] = __builtin_amdgcn_wmma_f32_16x16x32_bf16(
                    false, a[i], false, bq[j], (short)0, acc[i][j], false, false);
        buf ^= 1;
    }

    // Epilogue: C layout lane<16 -> M=v, N=lane ; lane>=16 -> M=8+v, N=lane-16
    #pragma unroll
    for (int i = 0; i < 4; ++i) {
        #pragma unroll
        for (int j = 0; j < 4; ++j) {
            const int col = colBase + (waveN * 4 + j) * 16 + ln16;
            const float bv = bias ? bias[col] : 0.0f;
            #pragma unroll
            for (int v = 0; v < 8; ++v) {
                const int row = rowBase + (waveM * 4 + i) * 16 + v + hi * 8;
                Out[(size_t)row * Nout + col] = acc[i][j][v] + bv;
            }
        }
    }
}

// ---------------------------------------------------------------- QKV LayerNorm
// qkv fp32 [B*N, 3*H*D].  One wave per (s,b,h,n) 64-element head row.
// s=0: LN*gamma+beta then *D^-0.5 -> Q bf16 [B,H,N,D]
// s=1: LN*gamma+beta              -> K bf16 [B,H,N,D]
// s=2: copy transposed            -> V^T bf16 [B,H,D,N]

__global__ __launch_bounds__(256)
void qkv_layernorm(const float* __restrict__ qkv,
                   const float* __restrict__ qg, const float* __restrict__ qb,
                   const float* __restrict__ kg, const float* __restrict__ kb,
                   u16* __restrict__ q_out, u16* __restrict__ k_out,
                   u16* __restrict__ vT_out, int B, int N)
{
    const int H = NUM_HEADS, D = HEAD_DIM;
    const int lane = threadIdx.x & 31;
    const int wid  = threadIdx.x >> 5;
    const long long r     = (long long)blockIdx.x * 8 + wid;
    const long long total = (long long)3 * B * H * N;
    if (r >= total) return;
    const int s  = (int)(r / ((long long)B * H * N));
    long long rem = r % ((long long)B * H * N);
    const int b  = (int)(rem / ((long long)H * N));
    const int hn = (int)(rem % ((long long)H * N));
    const int h  = hn / N;
    const int n  = hn % N;

    const float* src = qkv + ((size_t)b * N + n) * (3 * H * D) + (size_t)s * (H * D) + h * D;
    const int d0 = lane * 2;
    const float x0 = src[d0], x1 = src[d0 + 1];

    if (s < 2) {
        float sum = x0 + x1, sq = x0 * x0 + x1 * x1;
        #pragma unroll
        for (int m = 1; m < 32; m <<= 1) {
            sum += __shfl_xor(sum, m, 32);
            sq  += __shfl_xor(sq,  m, 32);
        }
        const float mu   = sum * (1.0f / D);
        const float var  = sq * (1.0f / D) - mu * mu;
        const float rstd = rsqrtf(var + LN_EPS);
        const float* g  = (s == 0) ? qg : kg;
        const float* be = (s == 0) ? qb : kb;
        const float sc  = (s == 0) ? 0.125f : 1.0f;          // D^-0.5 = 1/8
        const float y0 = ((x0 - mu) * rstd * g[d0]     + be[d0])     * sc;
        const float y1 = ((x1 - mu) * rstd * g[d0 + 1] + be[d0 + 1]) * sc;
        u16* dst = ((s == 0) ? q_out : k_out) + ((size_t)(b * H + h) * N + n) * D + d0;
        dst[0] = f2bf(y0); dst[1] = f2bf(y1);
    } else {
        u16* dst = vT_out + (size_t)(b * H + h) * D * N;
        dst[(size_t)d0 * N + n]       = f2bf(x0);
        dst[(size_t)(d0 + 1) * N + n] = f2bf(x1);
    }
}

// ---------------------------------------------------------------- Flash attention
// Block = 256 threads (8 waves), Q tile 128 rows (16 per wave), KV tiles of 64.
// K/V LDS double-buffered; next tile async-prefetched during compute.
// Q,K: [B,H,N,D] bf16 ; V^T: [B,H,D,N] bf16 ; out: [B,N,H*D] bf16.

__global__ __launch_bounds__(256)
void flash_attn(const u16* __restrict__ Q, const u16* __restrict__ Km,
                const u16* __restrict__ Vt, u16* __restrict__ Oo, int N)
{
    const int H = NUM_HEADS, D = HEAD_DIM;
    __shared__ __align__(16) u16 Kshm[2][2][4][32][16]; // [buf][dstep][nblk][lane][e]
    __shared__ __align__(16) u16 Vshm[2][2][4][32][16]; // [buf][kstep][dblk][lane][e]
    __shared__ __align__(16) u16 Pshm[8][16][72];       // per-wave P tile (pad 8 halves)

    const int tid  = threadIdx.x;
    const int lane = tid & 31;
    const int w    = tid >> 5;           // 0..7
    const int ln16 = lane & 15;
    const int hi   = (lane >> 4) & 1;

    const int qtile = blockIdx.x;
    const int h     = blockIdx.y;
    const int b     = blockIdx.z;

    const size_t headOff = ((size_t)b * H + h) * (size_t)N * D;
    const u16* Qh = Q  + headOff;
    const u16* Kh = Km + headOff;
    const u16* Vh = Vt + headOff;        // [D][N]

    const int qbase = qtile * 128;
    const int qrow  = qbase + w * 16 + ln16;

    v16bf aQ[2];
    #pragma unroll
    for (int ds = 0; ds < 2; ++ds) {     // A-frag: k = ds*32 + hi*16 + {0..15}
        const u16* p = Qh + (size_t)qrow * D + ds * 32 + hi * 16;
        aQ[ds] = frag_from_32B(p);
    }

    const v8f vzero = {0.f,0.f,0.f,0.f,0.f,0.f,0.f,0.f};
    v8f accO[4];
    #pragma unroll
    for (int d = 0; d < 4; ++d) accO[d] = vzero;
    float mrow[8], lrow[8];
    #pragma unroll
    for (int v = 0; v < 8; ++v) { mrow[v] = -1e30f; lrow[v] = 0.0f; }

    // Loader: 256 slots (dstep/kstep, blk, lane) == 256 threads -> 1 slot each.
    const int sl  = lane;
    const int snb = (tid >> 5) & 3;     // nblk for K, dblk for V
    const int sds = tid >> 7;           // dstep for K, kstep for V
    const int shi = (sl >> 4) & 1;

    auto issueKV = [&](int kbase, int buf) {
        {   // K fragment-ready: element = K[kcol][dstep*32 + hi16*16 + e]
            const int kcol = kbase + snb * 16 + (sl & 15);
            const u16* g = Kh + (size_t)kcol * D + sds * 32 + shi * 16;
            g2lds_16B(g,     &Kshm[buf][sds][snb][sl][0]);
            g2lds_16B(g + 8, &Kshm[buf][sds][snb][sl][8]);
        }
        {   // V fragment-ready: element = Vt[d][kbase + kstep*32 + hi16*16 + e]
            const int drow = snb * 16 + (sl & 15);
            const u16* g = Vh + (size_t)drow * N + kbase + sds * 32 + shi * 16;
            g2lds_16B(g,     &Vshm[buf][sds][snb][sl][0]);
            g2lds_16B(g + 8, &Vshm[buf][sds][snb][sl][8]);
        }
    };

    const int ntiles = N / 64;
    issueKV(0, 0);
    int buf = 0;
    for (int kt = 0; kt < ntiles; ++kt) {
        wait_g2lds();
        __syncthreads();
        if (kt + 1 < ntiles) issueKV((kt + 1) * 64, buf ^ 1);  // overlap with compute

        // S = Q * K^T : 4 column blocks of 16, K-dim = D = 64 (2 WMMAs each)
        v8f accS[4];
        #pragma unroll
        for (int nb = 0; nb < 4; ++nb) {
            v8f s = vzero;
            #pragma unroll
            for (int ds = 0; ds < 2; ++ds) {
                const v16bf bK = frag_from_32B(&Kshm[buf][ds][nb][lane][0]);
                s = __builtin_amdgcn_wmma_f32_16x16x32_bf16(
                        false, aQ[ds], false, bK, (short)0, s, false, false);
            }
            accS[nb] = s;
        }

        // Online softmax, per accumulator row v (row = v + 8*hi within 16-lane group)
        #pragma unroll
        for (int v = 0; v < 8; ++v) {
            float mx = fmaxf(fmaxf(accS[0][v], accS[1][v]),
                             fmaxf(accS[2][v], accS[3][v]));
            #pragma unroll
            for (int msk = 1; msk < 16; msk <<= 1) mx = fmaxf(mx, __shfl_xor(mx, msk, 32));
            const float mnew = fmaxf(mrow[v], mx);
            const float corr = __expf(mrow[v] - mnew);
            float rs = 0.0f;
            const int mr = v + hi * 8;
            #pragma unroll
            for (int nb = 0; nb < 4; ++nb) {
                const float p = __expf(accS[nb][v] - mnew);
                rs += p;
                Pshm[w][mr][nb * 16 + ln16] = f2bf(p);
            }
            #pragma unroll
            for (int msk = 1; msk < 16; msk <<= 1) rs += __shfl_xor(rs, msk, 32);
            lrow[v] = lrow[v] * corr + rs;
            mrow[v] = mnew;
            #pragma unroll
            for (int d = 0; d < 4; ++d) accO[d][v] *= corr;
        }

        // O += P * V.  Pshm is wave-private: same-wave LDS ops are in-order and
        // the compiler inserts the DScnt wait (may-alias), so no barrier needed.
        v16bf aP[2];
        #pragma unroll
        for (int ks = 0; ks < 2; ++ks) {
            const u16* p = &Pshm[w][ln16][ks * 32 + hi * 16];
            aP[ks] = frag_from_32B(p);
        }
        #pragma unroll
        for (int d = 0; d < 4; ++d)
            #pragma unroll
            for (int ks = 0; ks < 2; ++ks) {
                const v16bf bV = frag_from_32B(&Vshm[buf][ks][d][lane][0]);
                accO[d] = __builtin_amdgcn_wmma_f32_16x16x32_bf16(
                              false, aP[ks], false, bV, (short)0, accO[d], false, false);
            }
        buf ^= 1;
    }

    // Normalize and write out to [B, N, H*D] (bf16) for the projection GEMM.
    #pragma unroll
    for (int v = 0; v < 8; ++v) {
        const float rinv = 1.0f / lrow[v];
        const int row = qbase + w * 16 + v + hi * 8;
        #pragma unroll
        for (int d = 0; d < 4; ++d) {
            const int col = h * D + d * 16 + ln16;
            Oo[((size_t)b * N + row) * (H * D) + col] = f2bf(accO[d][v] * rinv);
        }
    }
}

// ---------------------------------------------------------------- launcher

extern "C" void kernel_launch(void* const* d_in, const int* in_sizes, int n_in,
                              void* d_out, int out_size, void* d_ws, size_t ws_size,
                              hipStream_t stream) {
    (void)in_sizes; (void)n_in; (void)out_size; (void)ws_size;
    const int B = 2, N = 4096, C = 768, H = NUM_HEADS, D = HEAD_DIM;
    const int M = B * N;       // 8192
    const int F = 3 * C;       // 2304

    const float* x      = (const float*)d_in[0];
    const float* qkv_w  = (const float*)d_in[1];
    const float* q_g    = (const float*)d_in[2];
    const float* q_b    = (const float*)d_in[3];
    const float* k_g    = (const float*)d_in[4];
    const float* k_b    = (const float*)d_in[5];
    const float* proj_w = (const float*)d_in[6];
    const float* proj_b = (const float*)d_in[7];
    float* out = (float*)d_out;

    char* ws = (char*)d_ws;
    size_t off = 0;
    u16*   x_bf    = (u16*)(ws + off);  off += (size_t)M * C * 2;
    u16*   qkvw_bf = (u16*)(ws + off);  off += (size_t)F * C * 2;
    u16*   projw_bf= (u16*)(ws + off);  off += (size_t)C * C * 2;
    size_t qkv_off = off;
    float* qkv_f   = (float*)(ws + off); off += (size_t)M * F * 4;
    u16*   q_ws    = (u16*)(ws + off);  off += (size_t)B * H * N * D * 2;
    u16*   k_ws    = (u16*)(ws + off);  off += (size_t)B * H * N * D * 2;
    u16*   vT_ws   = (u16*)(ws + off);  off += (size_t)B * H * N * D * 2;
    u16*   attn_ws = (u16*)(ws + qkv_off);   // reuse fp32-QKV region (dead after LN)

    cvt_f32_bf16<<<512, 256, 0, stream>>>(x,      x_bf,     M * C);
    cvt_f32_bf16<<<512, 256, 0, stream>>>(qkv_w,  qkvw_bf,  F * C);
    cvt_f32_bf16<<<256, 256, 0, stream>>>(proj_w, projw_bf, C * C);

    gemm_bf16_wmma<<<dim3(F / 256, M / 128), 256, 0, stream>>>(
        x_bf, qkvw_bf, nullptr, qkv_f, M, C, F);

    const int rows = 3 * B * H * N;
    qkv_layernorm<<<(rows + 7) / 8, 256, 0, stream>>>(
        qkv_f, q_g, q_b, k_g, k_b, q_ws, k_ws, vT_ws, B, N);

    flash_attn<<<dim3(N / 128, H, B), 256, 0, stream>>>(q_ws, k_ws, vT_ws, attn_ws, N);

    gemm_bf16_wmma<<<dim3(C / 256, M / 128), 256, 0, stream>>>(
        attn_ws, projw_bf, proj_b, out, M, C, C);
}
